// DirRecMultiHorizonHead_56573309223343
// MI455X (gfx1250) — matrix operand order repaired and checked by reading, not compiled, not run on hardware
//
#include <hip/hip_runtime.h>
#include <math.h>

#define BATCH 65536
#define DIM   256
#define HID   128
#define PLEN  48
#define MROWS 64
#define NTHREADS 256
#define BSTRIDE 132   // padded f32 row stride for sBase (breaks bank conflicts)

typedef __attribute__((ext_vector_type(16))) __bf16 v16bf;
typedef __attribute__((ext_vector_type(2)))  __bf16 v2bf;
typedef __attribute__((ext_vector_type(8)))  float  v8f;

union BFrag { v16bf v; unsigned int u[8]; uint4 q[2]; };

__device__ __forceinline__ unsigned int pack2bf(float a, float b) {
#if __has_builtin(__builtin_amdgcn_cvt_pk_bf16_f32)
    v2bf p = __builtin_amdgcn_cvt_pk_bf16_f32(a, b);
    return __builtin_bit_cast(unsigned int, p);
#else
    union { v2bf h; unsigned int u; } r;
    r.h = v2bf{(__bf16)a, (__bf16)b};
    return r.u;
#endif
}

__device__ __forceinline__ float geluf(float x) {
    // exact gelu: 0.5*x*(1+erf(x/sqrt(2)))
    return 0.5f * x * (1.0f + erff(x * 0.70710678118654752f));
}

__global__ __launch_bounds__(NTHREADS)
void dirrec_kernel(const float* __restrict__ x,  const float* __restrict__ W1,
                   const float* __restrict__ b1, const float* __restrict__ W2,
                   const float* __restrict__ b2, const float* __restrict__ Wo,
                   const float* __restrict__ bo, float* __restrict__ out)
{
    __shared__ float sBase[MROWS * BSTRIDE];          // base, f32, padded rows
    __shared__ unsigned int sHA[4 * 4 * 32 * 8];      // packed bf16 A-fragments [rb][kc][lane][8 dwords]
    __shared__ float sPred[2][MROWS];
    __shared__ float sB1[HID], sB2[HID], sWo[HID], sW1l[HID];
    __shared__ float sBo;

    const int t    = threadIdx.x;
    const int lane = t & 31;
    const int w    = t >> 5;
    const int rb   = w & 3;          // row-block (16 rows)
    const int cg   = w >> 2;         // column group (0..1), 4 tiles of 16 cols each
    const int l15  = lane & 15;
    const int g    = lane >> 4;      // lane half
    const int row0 = blockIdx.x * MROWS;
    const int rowA = rb * 16 + l15;  // this lane's row for A-fragment staging

    // ---------------- small shared init ----------------
    if (t < HID) {
        sB1[t]  = b1[t];
        sB2[t]  = b2[t];
        sWo[t]  = Wo[t];                      // Wo is (128,1) flat
        sW1l[t] = W1[(size_t)DIM * HID + t];  // W1[D] row
    }
    if (t < MROWS) sPred[0][t] = 0.0f;
    if (t == 0)    sBo = bo[0];
    __syncthreads();

    // ---------------- base = x @ W1[:D] + b1 via bf16 WMMA ----------------
    v8f acc[4];
    #pragma unroll
    for (int i = 0; i < 4; ++i) acc[i] = v8f{0.f,0.f,0.f,0.f,0.f,0.f,0.f,0.f};

    #pragma unroll 1
    for (int kcg = 0; kcg < 2; ++kcg) {
        // stage x chunk (64 rows x 128 k) into packed A layout
        #pragma unroll
        for (int kcs = 0; kcs < 2; ++kcs) {
            const int kcL = 2 * cg + kcs;
            unsigned int pk[8];
            #pragma unroll
            for (int q = 0; q < 2; ++q) {
                const int kk0 = g * 16 + q * 8;
                const float* xp = x + (size_t)(row0 + rowA) * DIM + kcg * 128 + kcL * 32 + kk0;
                float4 a0 = ((const float4*)xp)[0];
                float4 a1 = ((const float4*)xp)[1];
                pk[q*4+0] = pack2bf(a0.x, a0.y);
                pk[q*4+1] = pack2bf(a0.z, a0.w);
                pk[q*4+2] = pack2bf(a1.x, a1.y);
                pk[q*4+3] = pack2bf(a1.z, a1.w);
            }
            uint4* dst = (uint4*)&sHA[((rb * 4 + kcL) * 32 + lane) * 8];
            dst[0] = make_uint4(pk[0], pk[1], pk[2], pk[3]);
            dst[1] = make_uint4(pk[4], pk[5], pk[6], pk[7]);
        }
        __syncthreads();
        #pragma unroll
        for (int kc = 0; kc < 4; ++kc) {
            BFrag af;
            const uint4* src = (const uint4*)&sHA[((rb * 4 + kc) * 32 + lane) * 8];
            af.q[0] = src[0]; af.q[1] = src[1];
            #pragma unroll
            for (int tt = 0; tt < 4; ++tt) {
                BFrag bf;
                const int n = cg * 64 + tt * 16 + l15;
                const int kbase = kcg * 128 + kc * 32 + (g << 4);
                #pragma unroll
                for (int s = 0; s < 8; ++s) {
                    float f0 = W1[(size_t)(kbase + 2*s)     * HID + n];
                    float f1 = W1[(size_t)(kbase + 2*s + 1) * HID + n];
                    bf.u[s] = pack2bf(f0, f1);
                }
                acc[tt] = __builtin_amdgcn_wmma_f32_16x16x32_bf16(
                    false, af.v, false, bf.v, (short)0, acc[tt], false, false);
            }
        }
        __syncthreads();
    }

    // store base (+b1) as f32 in LDS
    #pragma unroll
    for (int tt = 0; tt < 4; ++tt) {
        const int n = cg * 64 + tt * 16 + l15;
        const float bb = sB1[n];
        #pragma unroll
        for (int v = 0; v < 8; ++v) {
            const int m = rb * 16 + v + (g << 3);
            sBase[m * BSTRIDE + n] = acc[tt][v] + bb;
        }
    }

    // ---------------- preload W2 bf16 B-fragments into registers ----------------
    v16bf w2f[4][4];
    #pragma unroll
    for (int tt = 0; tt < 4; ++tt) {
        const int n = cg * 64 + tt * 16 + l15;
        #pragma unroll
        for (int kc = 0; kc < 4; ++kc) {
            BFrag bf;
            const int kbase = kc * 32 + (g << 4);
            #pragma unroll
            for (int s = 0; s < 8; ++s) {
                float f0 = W2[(size_t)(kbase + 2*s)     * HID + n];
                float f1 = W2[(size_t)(kbase + 2*s + 1) * HID + n];
                bf.u[s] = pack2bf(f0, f1);
            }
            w2f[tt][kc] = bf.v;
        }
    }
    __syncthreads();   // sBase writes visible to all

    // ---------------- 48-step recurrence ----------------
    #pragma unroll 1
    for (int step = 0; step < PLEN; ++step) {
        const int cur = step & 1;
        const int nxt = cur ^ 1;

        // phase A: write previous pred to out, init next-pred accumulator,
        //          compute h = gelu(base + pred*w1_last) packed bf16 into sHA
        if (t < MROWS) {
            float pp = sPred[cur][t];
            if (step > 0) out[(size_t)(row0 + t) * PLEN + (step - 1)] = pp;
            sPred[nxt][t] = sBo;
        }
        {
            const float pred = sPred[cur][rowA];
            #pragma unroll
            for (int kcs = 0; kcs < 2; ++kcs) {
                const int kc = 2 * cg + kcs;
                unsigned int pk[8];
                #pragma unroll
                for (int q = 0; q < 2; ++q) {
                    const int n0 = kc * 32 + g * 16 + q * 8;
                    const float4* bp = (const float4*)&sBase[rowA * BSTRIDE + n0];
                    float4 a0 = bp[0], a1 = bp[1];
                    const float4* wp = (const float4*)&sW1l[n0];
                    float4 u0 = wp[0], u1 = wp[1];
                    float h0 = geluf(a0.x + pred * u0.x);
                    float h1 = geluf(a0.y + pred * u0.y);
                    float h2 = geluf(a0.z + pred * u0.z);
                    float h3 = geluf(a0.w + pred * u0.w);
                    float h4 = geluf(a1.x + pred * u1.x);
                    float h5 = geluf(a1.y + pred * u1.y);
                    float h6 = geluf(a1.z + pred * u1.z);
                    float h7 = geluf(a1.w + pred * u1.w);
                    pk[q*4+0] = pack2bf(h0, h1);
                    pk[q*4+1] = pack2bf(h2, h3);
                    pk[q*4+2] = pack2bf(h4, h5);
                    pk[q*4+3] = pack2bf(h6, h7);
                }
                uint4* dst = (uint4*)&sHA[((rb * 4 + kc) * 32 + lane) * 8];
                dst[0] = make_uint4(pk[0], pk[1], pk[2], pk[3]);
                dst[1] = make_uint4(pk[4], pk[5], pk[6], pk[7]);
            }
        }
        __syncthreads();

        // phase B: G = h @ W2 via 16 WMMAs (W2 frags resident in registers)
        #pragma unroll
        for (int i = 0; i < 4; ++i) acc[i] = v8f{0.f,0.f,0.f,0.f,0.f,0.f,0.f,0.f};
        #pragma unroll
        for (int kc = 0; kc < 4; ++kc) {
            BFrag af;
            const uint4* src = (const uint4*)&sHA[((rb * 4 + kc) * 32 + lane) * 8];
            af.q[0] = src[0]; af.q[1] = src[1];
            #pragma unroll
            for (int tt = 0; tt < 4; ++tt) {
                acc[tt] = __builtin_amdgcn_wmma_f32_16x16x32_bf16(
                    false, af.v, false, w2f[tt][kc], (short)0, acc[tt], false, false);
            }
        }

        // phase C: G = gelu(G + b2); pred = G @ wo + bo (shuffle-reduce + LDS atomics)
        float p[8];
        #pragma unroll
        for (int v = 0; v < 8; ++v) p[v] = 0.0f;
        #pragma unroll
        for (int tt = 0; tt < 4; ++tt) {
            const int n = cg * 64 + tt * 16 + l15;
            const float bb = sB2[n];
            const float ww = sWo[n];
            #pragma unroll
            for (int v = 0; v < 8; ++v) {
                float gg = geluf(acc[tt][v] + bb);
                p[v] += gg * ww;
            }
        }
        #pragma unroll
        for (int v = 0; v < 8; ++v) {
            p[v] += __shfl_xor(p[v], 1, 32);
            p[v] += __shfl_xor(p[v], 2, 32);
            p[v] += __shfl_xor(p[v], 4, 32);
            p[v] += __shfl_xor(p[v], 8, 32);
        }
        if (l15 == 0) {
            #pragma unroll
            for (int v = 0; v < 8; ++v)
                atomicAdd(&sPred[nxt][rb * 16 + v + (g << 3)], p[v]);
        }
        __syncthreads();
    }

    // final horizon output (step 47 result lives in sPred[0])
    if (t < MROWS) out[(size_t)(row0 + t) * PLEN + (PLEN - 1)] = sPred[0][t];
}

extern "C" void kernel_launch(void* const* d_in, const int* in_sizes, int n_in,
                              void* d_out, int out_size, void* d_ws, size_t ws_size,
                              hipStream_t stream) {
    const float* x  = (const float*)d_in[0];
    const float* W1 = (const float*)d_in[1];
    const float* b1 = (const float*)d_in[2];
    const float* W2 = (const float*)d_in[3];
    const float* b2 = (const float*)d_in[4];
    const float* Wo = (const float*)d_in[5];
    const float* bo = (const float*)d_in[6];
    float* out = (float*)d_out;

    dim3 grid(BATCH / MROWS);
    dim3 block(NTHREADS);
    hipLaunchKernelGGL(dirrec_kernel, grid, block, 0, stream,
                       x, W1, b1, W2, b2, Wo, bo, out);
}
